// Block_31954556682442
// MI455X (gfx1250) — matrix operand order repaired and checked by reading, not compiled, not run on hardware
//
#include <hip/hip_runtime.h>

// ============================================================================
// SpikFormer MoE-SSA + MLP block for MI455X (gfx1250, wave32, WMMA).
// Heavy GEMMs: v_wmma_f32_16x16x32_bf16 (f32 accum), LDS tiles stored in
// WMMA *fragment-major* order so fragment reads are 2x ds_load_b128/lane.
// K-contiguous operands staged with global_load_async_to_lds_b128 (ASYNCcnt);
// N-contiguous B staged with coalesced uint4 loads + ds scatter.
// Spike tensors (LIF outputs) are exact in bf16; attn counts (<=96) and
// router-mixed y counts (<=4) are also exact in bf16.
// ============================================================================

typedef __bf16 bf16_t;
typedef __attribute__((ext_vector_type(16))) __bf16 v16bf;
typedef __attribute__((ext_vector_type(8)))  float  v8f;

#define T_STEPS 4
#define BATCH   8
#define TBx     32
#define CDIM    384
#define NPIX    256
#define EDIM    96
#define NEXP    4
#define HIDF    2048
#define HHALF   1024
#define BNS     0.9999950000374997f   // 1/sqrt(1+1e-5)

// ---------------------------------------------------------------------------
// CDNA5 async global->LDS copies (GV mode) + ASYNCcnt wait.
// VDST = LDS byte offset (low 32 bits of flat LDS address), VADDR = 64-bit VA.
// ---------------------------------------------------------------------------
static __device__ __forceinline__ void async_copy_b128(void* lds_ptr, const void* gptr) {
  unsigned int       loff = (unsigned int)(unsigned long long)(uintptr_t)lds_ptr;
  unsigned long long ga   = (unsigned long long)(uintptr_t)gptr;
  asm volatile("global_load_async_to_lds_b128 %0, %1, off"
               :: "v"(loff), "v"(ga) : "memory");
}
static __device__ __forceinline__ void wait_async() {
  asm volatile("s_wait_asynccnt 0" ::: "memory");
}

// ---------------------------------------------------------------------------
// Fragment-major LDS index maps (CDNA5 ISA 7.12.2 16-bit layouts).
// A 16x32 subtile: lane = ((k>>3)&1)*16 + (m&15); within-lane jj has
//   k = ((jj>>3)<<4) | (lane>>4)<<3 | (jj&7)  => jj = ((k>>4)<<3) | (k&7).
// B 32x16 subtile: lane = ((k>>3)&1)*16 + (n&15); j = ((k>>4)<<3) | (k&7).
// Each lane's 16 halfs are contiguous (32B) -> 2x ds_load_b128 per fragment,
// and a lane's jj-run covers 8 contiguous K -> async b128 staging when the
// global operand is K-contiguous.
// ---------------------------------------------------------------------------
static __device__ __forceinline__ int a_lds_idx(int m, int k) {
  int lane = (((k >> 3) & 1) << 4) | (m & 15);
  int jj   = ((k >> 4) << 3) | (k & 7);
  return ((m >> 4) << 9) + (lane << 4) + jj;
}
static __device__ __forceinline__ int b_lds_idx(int k, int n) {
  int lane = (((k >> 3) & 1) << 4) | (n & 15);
  return ((n >> 4) << 9) + (lane << 4) + (((k >> 4) << 3) | (k & 7));
}

// ---------------------------------------------------------------------------
// LIF step: mem += (x - mem)/tau (tau=2); spike = H(mem - 1); hard reset.
// ---------------------------------------------------------------------------
static __device__ __forceinline__ float lif_step(float& mem, float x) {
  mem = 0.5f * (mem + x);
  float s = (mem >= 1.0f) ? 1.0f : 0.0f;
  mem *= (1.0f - s);
  return s;
}

// ---------------------------------------------------------------------------
// f32 -> bf16 conversion
// ---------------------------------------------------------------------------
__global__ void k_cvt_bf16(const float* __restrict__ in, bf16_t* __restrict__ out, long n) {
  long i = (long)blockIdx.x * blockDim.x + threadIdx.x;
  if (i < n) out[i] = (bf16_t)in[i];
}

// ---------------------------------------------------------------------------
// BN-affine folding: scale = g*BNS ; shift = bias*scale + beta
// ---------------------------------------------------------------------------
__global__ void k_affine(const float* __restrict__ g, const float* __restrict__ bias,
                         const float* __restrict__ be,
                         float* __restrict__ scale, float* __restrict__ shift, int n) {
  int i = blockIdx.x * blockDim.x + threadIdx.x;
  if (i < n) {
    float s = g[i] * BNS;
    scale[i] = s;
    shift[i] = (bias ? bias[i] * s : 0.0f) + be[i];
  }
}

// ---------------------------------------------------------------------------
// Generic batched WMMA GEMM:  C[bz] (MxN) = A[bz/a_bdiv] (MxK) * B[bz%b_bmod] (KxN)
// Block = 8 wave32; block tile 32(M) x 128(N); each wave owns a 16x32 C slab
// (two 16x16 f32 accumulators, A-fragment reused across both WMMAs).
// ---------------------------------------------------------------------------
__global__ void __launch_bounds__(256)
k_gemm_wmma(const bf16_t* __restrict__ A, const bf16_t* __restrict__ Bm,
            void* __restrict__ Cout,
            int M, int N, int K,
            long a_rs, long a_cs, long a_bstride, int a_bdiv,
            long b_rs, long b_cs, long b_bstride, int b_bmod,
            const float* __restrict__ scale, const float* __restrict__ shift,
            int aff_bstride, int out_bf16)
{
  __shared__ __align__(32) bf16_t Atile[2 * 512];    // 2 x (16x32) fragment-major
  __shared__ __align__(32) bf16_t Btile[8 * 512];    // 8 x (32x16) fragment-major

  int bz    = blockIdx.z;
  int abidx = bz / a_bdiv;
  const bf16_t* Ab = A  + (long)abidx * a_bstride + (long)(blockIdx.y * 32)  * a_rs;
  const bf16_t* Bb = Bm + (long)(bz % b_bmod) * b_bstride + (long)(blockIdx.x * 128) * b_cs;

  int tid  = threadIdx.y * 32 + threadIdx.x;
  int wave = threadIdx.y;
  int lane = threadIdx.x;
  int lm = lane & 15;        // C col within tile
  int lh = lane >> 4;        // lane-half selector (C row group)
  int wm = wave & 1;         // m-tile within block (0..1)
  int wn = wave >> 1;        // 32-col slab (0..3)
  int nloc = wn * 32;

  v8f acc0 = {0.f, 0.f, 0.f, 0.f, 0.f, 0.f, 0.f, 0.f};
  v8f acc1 = {0.f, 0.f, 0.f, 0.f, 0.f, 0.f, 0.f, 0.f};

  for (int k0 = 0; k0 < K; k0 += 32) {
    const bf16_t* At = Ab + (long)k0 * a_cs;
    const bf16_t* Bt = Bb + (long)k0 * b_rs;
    bool used_async = false;

    // ---- stage A tile (32 M x 32 K) into fragment-major LDS ----
    if (a_cs == 1) {
      // K-contiguous: one async b128 (8 halfs = 8 contiguous K) per chunk.
      used_async = true;
      if (tid < 128) {                       // 128 chunks
        int m = tid >> 2, seg = tid & 3;
        int k = seg * 8;                     // jj-run start maps to k-run start
        async_copy_b128(&Atile[a_lds_idx(m, k)], At + (long)m * a_rs + k);
      }
    } else {
      // strided (e.g. q^T view, m-contiguous): coalesced scalar scatter.
#pragma unroll
      for (int i = 0; i < 4; ++i) {          // 1024 elements
        int e = tid + i * 256;
        int k = e >> 5, m = e & 31;          // m fastest -> coalesced when a_rs==1
        Atile[a_lds_idx(m, k)] = At[(long)m * a_rs + (long)k * a_cs];
      }
    }

    // ---- stage B tile (32 K x 128 N) into fragment-major LDS ----
    if (b_rs == 1) {
      // K-contiguous (v^T view): async b128 along K.
      used_async = true;
#pragma unroll
      for (int i = 0; i < 2; ++i) {          // 512 chunks
        int c = tid + i * 256;
        int n = c >> 2, seg = c & 3;
        int k = seg * 8;
        async_copy_b128(&Btile[b_lds_idx(k, n)], Bt + (long)n * b_cs + k);
      }
    } else {
      // N-contiguous (b_cs==1): coalesced uint4 global loads + ds scatter.
#pragma unroll
      for (int i = 0; i < 2; ++i) {          // 512 x 8-half chunks
        int c = tid + i * 256;
        int k = c >> 4, n8 = (c & 15) * 8;
        union { uint4 u; unsigned short s[8]; } uu;
        uu.u = *(const uint4*)(Bt + (long)k * b_rs + n8);
#pragma unroll
        for (int t2 = 0; t2 < 8; ++t2)
          Btile[b_lds_idx(k, n8 + t2)] = __builtin_bit_cast(bf16_t, uu.s[t2]);
      }
    }
    if (used_async) wait_async();
    __syncthreads();

    // ---- fragments: one 32B contiguous read per lane (2x ds_load_b128) ----
    v16bf afrag = *(const v16bf*)&Atile[(wm << 9) + (lane << 4)];
    v16bf b0    = *(const v16bf*)&Btile[((nloc >> 4) << 9) + (lane << 4)];
    v16bf b1    = *(const v16bf*)&Btile[(((nloc >> 4) + 1) << 9) + (lane << 4)];

    acc0 = __builtin_amdgcn_wmma_f32_16x16x32_bf16(false, afrag, false, b0,
                                                   (short)0, acc0, false, false);
    acc1 = __builtin_amdgcn_wmma_f32_16x16x32_bf16(false, afrag, false, b1,
                                                   (short)0, acc1, false, false);
    __syncthreads();
  }

  // ---- epilogue: C 16x16 f32 layout -> row m0+r+lh*8, col n0+lm ----
  int m0 = blockIdx.y * 32 + wm * 16;
  int n0 = blockIdx.x * 128 + nloc;
  long cbase = (long)bz * M * N;
#pragma unroll
  for (int r = 0; r < 8; ++r) {
    int m = m0 + r + lh * 8;
    float v0 = acc0[r], v1 = acc1[r];
    if (scale) {
      int ai = aff_bstride * abidx + m;
      float sc = scale[ai], sh = shift[ai];
      v0 = v0 * sc + sh;
      v1 = v1 * sc + sh;
    }
    long ci = cbase + (long)m * N + n0 + lm;
    if (out_bf16) {
      ((bf16_t*)Cout)[ci]      = (bf16_t)v0;
      ((bf16_t*)Cout)[ci + 16] = (bf16_t)v1;
    } else {
      ((float*)Cout)[ci]      = v0;
      ((float*)Cout)[ci + 16] = v1;
    }
  }
}

// ---------------------------------------------------------------------------
// Generic LIF over layout (outer, T, B, inner): pre f32 -> spikes bf16
// ---------------------------------------------------------------------------
__global__ void k_lif(const float* __restrict__ pre, bf16_t* __restrict__ spk,
                      long inner, int nouter) {
  long total = (long)nouter * BATCH * inner;
  long idx = (long)blockIdx.x * blockDim.x + threadIdx.x;
  if (idx >= total) return;
  long o = idx / (BATCH * inner);
  long r = idx - o * (BATCH * inner);
  long b = r / inner;
  long i = r - b * inner;
  long obase = o * (long)T_STEPS * BATCH * inner;
  float mem = 0.0f;
#pragma unroll
  for (int t = 0; t < T_STEPS; ++t) {
    long a = obase + ((long)t * BATCH + b) * inner + i;
    spk[a] = (bf16_t)lif_step(mem, pre[a]);
  }
}

// ---------------------------------------------------------------------------
// Router: w = lif(bn(router_w @ xs + rb)) — tiny (E=4), plain VALU dot.
// wspk layout (TB, E, N) f32.
// ---------------------------------------------------------------------------
__global__ void k_router(const float* __restrict__ x, const float* __restrict__ rw,
                         const float* __restrict__ rb, const float* __restrict__ rg,
                         const float* __restrict__ rbe, float* __restrict__ wspk) {
  int idx = blockIdx.x * blockDim.x + threadIdx.x;
  if (idx >= BATCH * NEXP * NPIX) return;
  int n = idx % NPIX;
  int e = (idx / NPIX) % NEXP;
  int b = idx / (NPIX * NEXP);
  float sc = rg[e] * BNS;
  float sh = rb[e] * sc + rbe[e];
  float mem = 0.0f;
  for (int t = 0; t < T_STEPS; ++t) {
    const float* xb = x + (long)(t * BATCH + b) * CDIM * NPIX + n;
    float acc = 0.0f;
    for (int c = 0; c < CDIM; ++c) acc += rw[e * CDIM + c] * xb[(long)c * NPIX];
    float s = lif_step(mem, acc * sc + sh);
    wspk[((long)(t * BATCH + b) * NEXP + e) * NPIX + n] = s;
  }
}

// ---------------------------------------------------------------------------
// y[tb,d,n] = sum_e w[tb,e,n] * res_spk[e,tb,n,d]  (integer 0..4 -> bf16 exact)
// ---------------------------------------------------------------------------
__global__ void k_yacc(const float* __restrict__ wspk, const bf16_t* __restrict__ rspk,
                       bf16_t* __restrict__ ybf) {
  long idx = (long)blockIdx.x * blockDim.x + threadIdx.x;
  if (idx >= (long)TBx * CDIM * NPIX) return;
  int n  = idx % NPIX;
  int d  = (idx / NPIX) % CDIM;
  int tb = idx / ((long)NPIX * CDIM);
  float acc = 0.0f;
#pragma unroll
  for (int e = 0; e < NEXP; ++e) {
    float w = wspk[((long)tb * NEXP + e) * NPIX + n];
    float r = (float)rspk[(((long)e * TBx + tb) * NPIX + n) * CDIM + d];
    acc += w * r;
  }
  ybf[idx] = (bf16_t)acc;   // (TB, C, N)
}

// ---------------------------------------------------------------------------
// proj LIF + residual: xres = x + spike(lif(ppre)); also bf16 copy for fc1.
// ---------------------------------------------------------------------------
__global__ void k_lif_res(const float* __restrict__ pre, const float* __restrict__ x,
                          float* __restrict__ xres, bf16_t* __restrict__ xrbf) {
  const long inner = (long)CDIM * NPIX;
  long idx = (long)blockIdx.x * blockDim.x + threadIdx.x;
  if (idx >= (long)BATCH * inner) return;
  long b = idx / inner, i = idx - b * inner;
  float mem = 0.0f;
#pragma unroll
  for (int t = 0; t < T_STEPS; ++t) {
    long a = ((long)t * BATCH + b) * inner + i;
    float o = x[a] + lif_step(mem, pre[a]);
    xres[a] = o;
    xrbf[a] = (bf16_t)o;
  }
}

// ---------------------------------------------------------------------------
// Depthwise 3x3 SAME on x1 spikes (first HHALF channels of hspk) + folded BN.
// ---------------------------------------------------------------------------
__global__ void k_dwconv(const bf16_t* __restrict__ hspk, const float* __restrict__ dww,
                         const float* __restrict__ dwb, const float* __restrict__ dwg,
                         const float* __restrict__ dwbe, float* __restrict__ dpre) {
  long idx = (long)blockIdx.x * blockDim.x + threadIdx.x;
  if (idx >= (long)TBx * HHALF * NPIX) return;
  int p  = idx % NPIX;
  int c  = (idx / NPIX) % HHALF;
  int tb = idx / ((long)NPIX * HHALF);
  int py = p >> 4, px = p & 15;
  const bf16_t* xin = hspk + ((long)tb * HIDF + c) * NPIX;
  float acc = 0.0f;
#pragma unroll
  for (int ky = 0; ky < 3; ++ky) {
    int iy = py + ky - 1;
    if (iy < 0 || iy >= 16) continue;
#pragma unroll
    for (int kx = 0; kx < 3; ++kx) {
      int ix = px + kx - 1;
      if (ix < 0 || ix >= 16) continue;
      acc += (float)xin[iy * 16 + ix] * dww[c * 9 + ky * 3 + kx];
    }
  }
  float sc = dwg[c] * BNS;
  dpre[idx] = acc * sc + (dwb[c] * sc + dwbe[c]);
}

// ---------------------------------------------------------------------------
// dw LIF then gate with x2 spikes: m = spike * x2  (0/1 -> bf16 exact)
// ---------------------------------------------------------------------------
__global__ void k_lif_gate(const float* __restrict__ dpre, const bf16_t* __restrict__ hspk,
                           bf16_t* __restrict__ mbf) {
  const long inner = (long)HHALF * NPIX;
  long idx = (long)blockIdx.x * blockDim.x + threadIdx.x;
  if (idx >= (long)BATCH * inner) return;
  long b = idx / inner, i = idx - b * inner;
  int c = i / NPIX, p = i % NPIX;
  float mem = 0.0f;
#pragma unroll
  for (int t = 0; t < T_STEPS; ++t) {
    int tb = t * BATCH + b;
    long a = ((long)tb * HHALF + c) * NPIX + p;
    float s = lif_step(mem, dpre[a]);
    float x2 = (float)hspk[((long)tb * HIDF + HHALF + c) * NPIX + p];
    mbf[a] = (bf16_t)(s * x2);
  }
}

// ---------------------------------------------------------------------------
// fc2 LIF + final residual: out = xres + spike
// ---------------------------------------------------------------------------
__global__ void k_lif_final(const float* __restrict__ pre, const float* __restrict__ xres,
                            float* __restrict__ out) {
  const long inner = (long)CDIM * NPIX;
  long idx = (long)blockIdx.x * blockDim.x + threadIdx.x;
  if (idx >= (long)BATCH * inner) return;
  long b = idx / inner, i = idx - b * inner;
  float mem = 0.0f;
#pragma unroll
  for (int t = 0; t < T_STEPS; ++t) {
    long a = ((long)t * BATCH + b) * inner + i;
    out[a] = xres[a] + lif_step(mem, pre[a]);
  }
}

// ============================================================================
// Workspace layout (static bump allocation; 95MB arena aliased across phases)
// ============================================================================
constexpr size_t AL(size_t x) { return (x + 255) & ~(size_t)255; }
constexpr size_t O_KW   = 0;
constexpr size_t O_VW   = AL(O_KW  + (size_t)EDIM * CDIM * 2);
constexpr size_t O_EW   = AL(O_VW  + (size_t)CDIM * CDIM * 2);
constexpr size_t O_PW   = AL(O_EW  + (size_t)NEXP * EDIM * CDIM * 2);
constexpr size_t O_F1W  = AL(O_PW  + (size_t)CDIM * CDIM * 2);
constexpr size_t O_F2W  = AL(O_F1W + (size_t)HIDF * CDIM * 2);
constexpr size_t O_AQS  = AL(O_F2W + (size_t)CDIM * HHALF * 2);
constexpr size_t O_AQH  = AL(O_AQS + (size_t)NEXP * EDIM * 4);
constexpr size_t O_APS  = AL(O_AQH + (size_t)NEXP * EDIM * 4);
constexpr size_t O_APH  = AL(O_APS + (size_t)CDIM * 4);
constexpr size_t O_A1S  = AL(O_APH + (size_t)CDIM * 4);
constexpr size_t O_A1H  = AL(O_A1S + (size_t)HIDF * 4);
constexpr size_t O_A2S  = AL(O_A1H + (size_t)HIDF * 4);
constexpr size_t O_A2H  = AL(O_A2S + (size_t)CDIM * 4);
constexpr size_t O_XSBF = AL(O_A2H + (size_t)CDIM * 4);
constexpr size_t O_KSPK = AL(O_XSBF + (size_t)TBx * CDIM * NPIX * 2);
constexpr size_t O_VSPK = AL(O_KSPK + (size_t)TBx * EDIM * NPIX * 2);
constexpr size_t O_QSPK = AL(O_VSPK + (size_t)TBx * CDIM * NPIX * 2);
constexpr size_t O_WSPK = AL(O_QSPK + (size_t)NEXP * TBx * EDIM * NPIX * 2);
constexpr size_t O_YBF  = AL(O_WSPK + (size_t)TBx * NEXP * NPIX * 4);
constexpr size_t O_HSPK = AL(O_YBF  + (size_t)TBx * CDIM * NPIX * 2);
constexpr size_t O_XRES = AL(O_HSPK + (size_t)TBx * HIDF * NPIX * 2);
constexpr size_t O_XRBF = AL(O_XRES + (size_t)TBx * CDIM * NPIX * 4);
constexpr size_t O_RSPK = AL(O_XRBF + (size_t)TBx * CDIM * NPIX * 2);
constexpr size_t O_MBF  = O_RSPK;  // alias: rspk dead before gate phase
constexpr size_t O_ARENA= AL(O_RSPK + (size_t)NEXP * TBx * NPIX * CDIM * 2);
// arena phase 1 (SSA pre-activations)
constexpr size_t O_KPRE = O_ARENA;
constexpr size_t O_VPRE = AL(O_KPRE + (size_t)TBx * EDIM * NPIX * 4);
constexpr size_t O_QPRE = AL(O_VPRE + (size_t)TBx * CDIM * NPIX * 4);
constexpr size_t O_ATTN = AL(O_QPRE + (size_t)NEXP * TBx * EDIM * NPIX * 4);
constexpr size_t O_RPRE = AL(O_ATTN + (size_t)NEXP * TBx * NPIX * NPIX * 2);
// arena aliases (earlier buffers dead by the time each is written)
constexpr size_t O_PPRE  = O_ARENA;                                  // proj pre
constexpr size_t O_HPRE  = O_ARENA;                                  // fc1 pre (67MB)
constexpr size_t O_DPRE  = O_ARENA;                                  // dw pre (34MB)
constexpr size_t O_F2PRE = AL(O_ARENA + (size_t)TBx * HHALF * NPIX * 4);

extern "C" void kernel_launch(void* const* d_in, const int* in_sizes, int n_in,
                              void* d_out, int out_size, void* d_ws, size_t ws_size,
                              hipStream_t stream) {
  const float* x        = (const float*)d_in[0];
  const float* k_w      = (const float*)d_in[2];
  const float* v_w      = (const float*)d_in[3];
  const float* router_w = (const float*)d_in[4];
  const float* router_b = (const float*)d_in[5];
  const float* router_g = (const float*)d_in[6];
  const float* router_be= (const float*)d_in[7];
  const float* exp_w    = (const float*)d_in[8];
  const float* exp_g    = (const float*)d_in[9];
  const float* exp_b    = (const float*)d_in[10];
  const float* proj_w   = (const float*)d_in[11];
  const float* proj_b   = (const float*)d_in[12];
  const float* proj_g   = (const float*)d_in[13];
  const float* proj_be  = (const float*)d_in[14];
  const float* fc1_w    = (const float*)d_in[15];
  const float* fc1_b    = (const float*)d_in[16];
  const float* fc1_g    = (const float*)d_in[17];
  const float* fc1_be   = (const float*)d_in[18];
  const float* dw_w     = (const float*)d_in[19];
  const float* dw_b     = (const float*)d_in[20];
  const float* dw_g     = (const float*)d_in[21];
  const float* dw_be    = (const float*)d_in[22];
  const float* fc2_w    = (const float*)d_in[23];
  const float* fc2_b    = (const float*)d_in[24];
  const float* fc2_g    = (const float*)d_in[25];
  const float* fc2_be   = (const float*)d_in[26];

  char* ws = (char*)d_ws;
  bf16_t* xsbf = (bf16_t*)(ws + O_XSBF);
  bf16_t* wkbf = (bf16_t*)(ws + O_KW);
  bf16_t* wvbf = (bf16_t*)(ws + O_VW);
  bf16_t* webf = (bf16_t*)(ws + O_EW);
  bf16_t* wpbf = (bf16_t*)(ws + O_PW);
  bf16_t* w1bf = (bf16_t*)(ws + O_F1W);
  bf16_t* w2bf = (bf16_t*)(ws + O_F2W);
  float*  aqs  = (float*)(ws + O_AQS); float* aqh = (float*)(ws + O_AQH);
  float*  aps  = (float*)(ws + O_APS); float* aph = (float*)(ws + O_APH);
  float*  a1s  = (float*)(ws + O_A1S); float* a1h = (float*)(ws + O_A1H);
  float*  a2s  = (float*)(ws + O_A2S); float* a2h = (float*)(ws + O_A2H);
  float*  kpre = (float*)(ws + O_KPRE);  bf16_t* kspk = (bf16_t*)(ws + O_KSPK);
  float*  vpre = (float*)(ws + O_VPRE);  bf16_t* vspk = (bf16_t*)(ws + O_VSPK);
  float*  qpre = (float*)(ws + O_QPRE);  bf16_t* qspk = (bf16_t*)(ws + O_QSPK);
  float*  wspk = (float*)(ws + O_WSPK);
  bf16_t* attn = (bf16_t*)(ws + O_ATTN);
  float*  rpre = (float*)(ws + O_RPRE);  bf16_t* rspk = (bf16_t*)(ws + O_RSPK);
  bf16_t* ybf  = (bf16_t*)(ws + O_YBF);
  float*  ppre = (float*)(ws + O_PPRE);
  float*  xres = (float*)(ws + O_XRES);  bf16_t* xrbf = (bf16_t*)(ws + O_XRBF);
  float*  hpre = (float*)(ws + O_HPRE);  bf16_t* hspk = (bf16_t*)(ws + O_HSPK);
  float*  dpre = (float*)(ws + O_DPRE);  bf16_t* mbf  = (bf16_t*)(ws + O_MBF);
  float*  f2pre= (float*)(ws + O_F2PRE);
  float*  out  = (float*)d_out;

  auto cvt = [&](const float* in, bf16_t* o, long n) {
    k_cvt_bf16<<<dim3((unsigned)((n + 255) / 256)), dim3(256), 0, stream>>>(in, o, n);
  };

  // ---- Phase 0: bf16 packs + folded BN affines ----
  cvt(x,     xsbf, (long)TBx * CDIM * NPIX);
  cvt(k_w,   wkbf, (long)EDIM * CDIM);
  cvt(v_w,   wvbf, (long)CDIM * CDIM);
  cvt(exp_w, webf, (long)NEXP * EDIM * CDIM);
  cvt(proj_w,wpbf, (long)CDIM * CDIM);
  cvt(fc1_w, w1bf, (long)HIDF * CDIM);
  cvt(fc2_w, w2bf, (long)CDIM * HHALF);
  k_affine<<<dim3(2), dim3(256), 0, stream>>>(exp_g,  nullptr, exp_b,   aqs, aqh, NEXP * EDIM);
  k_affine<<<dim3(2), dim3(256), 0, stream>>>(proj_g, proj_b,  proj_be, aps, aph, CDIM);
  k_affine<<<dim3(8), dim3(256), 0, stream>>>(fc1_g,  fc1_b,   fc1_be,  a1s, a1h, HIDF);
  k_affine<<<dim3(2), dim3(256), 0, stream>>>(fc2_g,  fc2_b,   fc2_be,  a2s, a2h, CDIM);

  const dim3 blk(32, 8);
  const long XB = (long)CDIM * NPIX;      // per-batch xs stride

  // ---- Phase 1: k / v / q GEMMs + router + LIFs ----
  k_gemm_wmma<<<dim3(2, 3, 32), blk, 0, stream>>>(wkbf, xsbf, kpre, EDIM, NPIX, CDIM,
      CDIM, 1, 0, 1,  NPIX, 1, XB, TBx,  nullptr, nullptr, 0, 0);
  k_gemm_wmma<<<dim3(2, 12, 32), blk, 0, stream>>>(wvbf, xsbf, vpre, CDIM, NPIX, CDIM,
      CDIM, 1, 0, 1,  NPIX, 1, XB, TBx,  nullptr, nullptr, 0, 0);
  k_gemm_wmma<<<dim3(2, 3, 128), blk, 0, stream>>>(webf, xsbf, qpre, EDIM, NPIX, CDIM,
      CDIM, 1, (long)EDIM * CDIM, TBx,  NPIX, 1, XB, TBx,  aqs, aqh, EDIM, 0);
  k_router<<<dim3(32), dim3(256), 0, stream>>>(x, router_w, router_b, router_g, router_be, wspk);
  k_lif<<<dim3(768),  dim3(256), 0, stream>>>(kpre, kspk, (long)EDIM * NPIX, 1);
  k_lif<<<dim3(3072), dim3(256), 0, stream>>>(vpre, vspk, (long)CDIM * NPIX, 1);
  k_lif<<<dim3(3072), dim3(256), 0, stream>>>(qpre, qspk, (long)EDIM * NPIX, NEXP);

  // ---- Phase 2: attn = q^T k ; res = attn v^T ; LIF ; router mix ; proj ----
  k_gemm_wmma<<<dim3(2, 8, 128), blk, 0, stream>>>(qspk, kspk, attn, NPIX, NPIX, EDIM,
      1, NPIX, (long)EDIM * NPIX, 1,           // A = q^T view (per eb) -> scatter stage
      NPIX, 1, (long)EDIM * NPIX, TBx,         // B = k[b] -> uint4 + ds scatter
      nullptr, nullptr, 0, /*out_bf16=*/1);
  k_gemm_wmma<<<dim3(3, 8, 128), blk, 0, stream>>>(attn, vspk, rpre, NPIX, CDIM, NPIX,
      NPIX, 1, (long)NPIX * NPIX, 1,           // A = attn[eb] -> async b128 stage
      1, NPIX, (long)CDIM * NPIX, TBx,         // B = v^T view (per b) -> async b128 stage
      nullptr, nullptr, 0, 0);
  k_lif<<<dim3(12288),  dim3(256), 0, stream>>>(rpre, rspk, (long)NPIX * CDIM, NEXP);
  k_yacc<<<dim3(12288), dim3(256), 0, stream>>>(wspk, rspk, ybf);
  k_gemm_wmma<<<dim3(2, 12, 32), blk, 0, stream>>>(wpbf, ybf, ppre, CDIM, NPIX, CDIM,
      CDIM, 1, 0, 1,  NPIX, 1, XB, TBx,  aps, aph, 0, 0);
  k_lif_res<<<dim3(3072), dim3(256), 0, stream>>>(ppre, x, xres, xrbf);

  // ---- Phase 3: MLP ----
  k_gemm_wmma<<<dim3(2, 64, 32), blk, 0, stream>>>(w1bf, xrbf, hpre, HIDF, NPIX, CDIM,
      CDIM, 1, 0, 1,  NPIX, 1, XB, TBx,  a1s, a1h, 0, 0);
  k_lif<<<dim3(16384), dim3(256), 0, stream>>>(hpre, hspk, (long)HIDF * NPIX, 1);
  k_dwconv<<<dim3(32768), dim3(256), 0, stream>>>(hspk, dw_w, dw_b, dw_g, dw_be, dpre);
  k_lif_gate<<<dim3(8192), dim3(256), 0, stream>>>(dpre, hspk, mbf);
  k_gemm_wmma<<<dim3(2, 12, 32), blk, 0, stream>>>(w2bf, mbf, f2pre, CDIM, NPIX, HHALF,
      HHALF, 1, 0, 1,  NPIX, 1, (long)HHALF * NPIX, TBx,  a2s, a2h, 0, 0);
  k_lif_final<<<dim3(3072), dim3(256), 0, stream>>>(f2pre, xres, out);

  (void)in_sizes; (void)n_in; (void)out_size; (void)ws_size;
}
